// Net_8650064134239
// MI455X (gfx1250) — compile-verified, matrix operand order checked by reading, not run on hardware
//
#include <hip/hip_runtime.h>
#include <hip/hip_bf16.h>
#include <math.h>

typedef _Float16 v16h __attribute__((ext_vector_type(16)));
typedef _Float16 v8h  __attribute__((ext_vector_type(8)));
typedef float    v8f  __attribute__((ext_vector_type(8)));

#define NN      65536     // nodes
#define NE      524288    // edges
#define NG      64        // graphs
#define DEMB    128
#define NSTEPS  4

// fragment buffer geometry (halves)
#define FRAG_STEP_STRIDE 81920     // (4+12+4) kc * 8 nt * 512
#define FRAG_MESS_OFF    0
#define FRAG_AGG_OFF     16384
#define FRAG_FEAT_OFF    65536

// workspace byte offsets
static const size_t OFF_X     = 0;           // N*128 f32
static const size_t OFF_XH    = 33554432;    // N*128 f16
static const size_t OFF_AGG   = 50331648;    // N*128 f32
static const size_t OFF_XG    = 100663296;   // 2 * G*128 f32 (ping-pong)
static const size_t OFF_XGH   = 100728832;   // G*128 f16
static const size_t OFF_XGACC = 100745216;   // G*128 f32
static const size_t OFF_GATE  = 100777984;   // N f32
static const size_t OFF_GMAX  = 101040128;   // G f32
static const size_t OFF_GDEN  = 101040384;   // G f32
static const size_t OFF_NMAX  = 101040640;   // G*5 f32
static const size_t OFF_NSUM  = 101041920;   // G*5 f32
static const size_t OFF_NLOG  = 101043200;   // N*5 f32
static const size_t OFF_FRAG  = 102353920;   // 327680 f16 (32B aligned)

__device__ __forceinline__ float lrelu(float v) { return v > 0.0f ? v : 0.01f * v; }

__device__ __forceinline__ void atomicMaxF(float* a, float v) {
    if (v >= 0.0f) atomicMax((int*)a, __float_as_int(v));
    else           atomicMin((unsigned int*)a, __float_as_uint(v));
}

__device__ __forceinline__ v16h load_a_frag(const _Float16* pa) {
    v8h lo = *(const v8h*)pa;          // K = kbase .. kbase+7
    v8h hi = *(const v8h*)(pa + 16);   // K = kbase+16 .. kbase+23
    return __builtin_shufflevector(lo, hi, 0,1,2,3,4,5,6,7,8,9,10,11,12,13,14,15);
}

// CDNA5 async global->LDS copy of 32 bytes (2 x b128), tracked by ASYNCcnt.
// LDS address comes from vdst, global address from vaddr; the imm offset is added to both
__device__ __forceinline__ void async_copy32B(const _Float16* gp, _Float16* lp) {
    unsigned l = (unsigned)(uintptr_t)lp;   // low 32 bits of flat LDS ptr = LDS address
    asm volatile(
        "global_load_async_to_lds_b128 %0, %1, off\n\t"
        "global_load_async_to_lds_b128 %0, %1, off offset:16"
        :: "v"(l), "v"(gp) : "memory");
}
__device__ __forceinline__ void async_wait0() {
    asm volatile("s_wait_asynccnt 0" ::: "memory");
}

// ---------------- weight prep: f32 -> f16 fragments in WMMA B layout ----------------
__global__ void k_prep(const float* __restrict__ messW, const float* __restrict__ aggW,
                       const float* __restrict__ featW, _Float16* __restrict__ frag) {
    int id = blockIdx.x * 256 + threadIdx.x;       // < 327680
    int s = id / FRAG_STEP_STRIDE;
    int r = id % FRAG_STEP_STRIDE;
    const float* W; int local;
    if (r < 16384)       { W = messW + (size_t)s * 132 * 128; local = r; }
    else if (r < 65536)  { W = aggW  + (size_t)s * 384 * 128; local = r - 16384; }
    else                 { W = featW + (size_t)s * 128 * 128; local = r - 65536; }
    int kc = local >> 12;
    int r2 = local & 4095;
    int nt = r2 >> 9;
    int r3 = r2 & 511;
    int lane = r3 >> 4;
    int h = r3 & 15;
    int hw = lane >> 4, nn = lane & 15;
    int kl = (h < 8) ? (hw * 8 + h) : (16 + hw * 8 + (h - 8));
    int k = kc * 32 + kl, n = nt * 16 + nn;
    frag[id] = (_Float16)W[(size_t)k * 128 + n];
}

__global__ void k_zero_xg(float* xg) { xg[blockIdx.x * 256 + threadIdx.x] = 0.0f; }

// ---------------- embed: x = lrelu(nf @ W + b) ----------------
__global__ void k_embed(const float* __restrict__ nf, const float* __restrict__ W,
                        const float* __restrict__ b, float* __restrict__ x,
                        _Float16* __restrict__ xh) {
    int n = blockIdx.x, c = threadIdx.x;
    float f0 = nf[n*5+0], f1 = nf[n*5+1], f2 = nf[n*5+2], f3 = nf[n*5+3], f4 = nf[n*5+4];
    float v = b[c] + f0*W[c] + f1*W[128+c] + f2*W[256+c] + f3*W[384+c] + f4*W[512+c];
    v = lrelu(v);
    x[(size_t)n*128 + c] = v;
    xh[(size_t)n*128 + c] = (_Float16)v;
}

// ---------------- per-step init ----------------
__global__ void k_step_init(float* __restrict__ agg, float* __restrict__ xgacc,
                            _Float16* __restrict__ xgh, const float* __restrict__ xgcur,
                            float* __restrict__ gmax, float* __restrict__ gden) {
    size_t tid = (size_t)blockIdx.x * 256 + threadIdx.x;
    agg[tid] = -__builtin_inff();
    if (tid < NG * DEMB) { xgacc[tid] = 0.0f; xgh[tid] = (_Float16)xgcur[tid]; }
    if (tid < NG)        { gmax[tid] = -__builtin_inff(); gden[tid] = 0.0f; }
}

// ---------------- edge GEMM + scatter-max (WMMA) ----------------
__global__ void __launch_bounds__(128) k_message(
    const _Float16* __restrict__ xh, const float* __restrict__ ea,
    const int* __restrict__ src, const int* __restrict__ dst,
    const _Float16* __restrict__ wfrag, const float* __restrict__ we,
    const float* __restrict__ bias, float* __restrict__ agg) {
    __shared__ __align__(16) _Float16 lx[16 * 136];
    __shared__ float lea[64];
    __shared__ int lsrc[16], ldst[16];
    int t = threadIdx.x;
    size_t e0 = (size_t)blockIdx.x * 16;
    if (t < 16) { lsrc[t] = src[e0 + t]; ldst[t] = dst[e0 + t]; }
    if (t >= 64) lea[t - 64] = ea[e0 * 4 + (t - 64)];
    if (t == 32) __builtin_prefetch(src + e0 + 16, 0, 0);   // global_prefetch_b8
    __syncthreads();
    {   // async gather of 16 source rows into LDS (32B per thread)
        int row = t >> 3, seg = t & 7;
        async_copy32B(xh + (size_t)lsrc[row] * 128 + seg * 16,
                      &lx[row * 136 + seg * 16]);
    }
    async_wait0();
    __syncthreads();
    int lane = t & 31, wv = t >> 5, hw = lane >> 4, nn = lane & 15;
    v8f c[2] = {v8f{}, v8f{}};
#pragma unroll
    for (int kc = 0; kc < 4; ++kc) {
        v16h a = load_a_frag(&lx[nn * 136 + kc * 32 + hw * 8]);
#pragma unroll
        for (int j = 0; j < 2; ++j) {
            const v16h* pb = (const v16h*)(wfrag + (size_t)(kc * 8 + wv * 2 + j) * 512);
            v16h b = pb[lane];
            c[j] = __builtin_amdgcn_wmma_f32_16x16x32_f16(false, a, false, b,
                                                          (short)0, c[j], false, false);
        }
    }
#pragma unroll
    for (int j = 0; j < 2; ++j) {
        int col = (wv * 2 + j) * 16 + nn;
        float w0 = we[col], w1 = we[128 + col], w2 = we[256 + col], w3 = we[384 + col];
        float bv = bias[col];
#pragma unroll
        for (int r = 0; r < 8; ++r) {
            int mrow = hw * 8 + r;
            float v = c[j][r] + bv + lea[mrow*4]*w0 + lea[mrow*4+1]*w1
                    + lea[mrow*4+2]*w2 + lea[mrow*4+3]*w3;
            atomicMaxF(&agg[(size_t)ldst[mrow] * 128 + col], lrelu(v));
        }
    }
}

// ---------------- node update GEMM, K=384 (WMMA); agg fix folded into gather ----------------
__global__ void __launch_bounds__(128) k_update(
    _Float16* __restrict__ xh, const _Float16* __restrict__ xgh,
    const float* __restrict__ agg, const _Float16* __restrict__ wfrag,
    const float* __restrict__ bias, float* __restrict__ x) {
    __shared__ __align__(16) _Float16 lx[16 * 392];
    int t = threadIdx.x;
    int n0 = blockIdx.x * 16;
    {
        int row = t >> 3;
        int node = n0 + row;
        int g = node >> 10;
        int c0 = (t & 7) * 16;
        // cols [0,128): x_half  -- async global->LDS
        async_copy32B(xh + (size_t)node * 128 + c0, &lx[row * 392 + c0]);
        // cols [128,256): xg_half broadcast -- async global->LDS
        async_copy32B(xgh + (size_t)g * 128 + c0, &lx[row * 392 + 128 + c0]);
        // cols [256,384): agg f32 -> fix(-inf->0) -> f16 -> LDS
        const float4* ap = (const float4*)(agg + (size_t)node * 128 + c0);
        _Float16* q = &lx[row * 392 + 256 + c0];
#pragma unroll
        for (int u = 0; u < 2; ++u) {
            float4 va = ap[u * 2], vb = ap[u * 2 + 1];
            v8h h;
            h[0] = (_Float16)(va.x > -__builtin_inff() ? va.x : 0.0f);
            h[1] = (_Float16)(va.y > -__builtin_inff() ? va.y : 0.0f);
            h[2] = (_Float16)(va.z > -__builtin_inff() ? va.z : 0.0f);
            h[3] = (_Float16)(va.w > -__builtin_inff() ? va.w : 0.0f);
            h[4] = (_Float16)(vb.x > -__builtin_inff() ? vb.x : 0.0f);
            h[5] = (_Float16)(vb.y > -__builtin_inff() ? vb.y : 0.0f);
            h[6] = (_Float16)(vb.z > -__builtin_inff() ? vb.z : 0.0f);
            h[7] = (_Float16)(vb.w > -__builtin_inff() ? vb.w : 0.0f);
            *(v8h*)(q + u * 8) = h;
        }
    }
    async_wait0();
    __syncthreads();
    int lane = t & 31, wv = t >> 5, hw = lane >> 4, nn = lane & 15;
    v8f c[2] = {v8f{}, v8f{}};
#pragma unroll
    for (int kc = 0; kc < 12; ++kc) {
        v16h a = load_a_frag(&lx[nn * 392 + kc * 32 + hw * 8]);
#pragma unroll
        for (int j = 0; j < 2; ++j) {
            const v16h* pb = (const v16h*)(wfrag + (size_t)(kc * 8 + wv * 2 + j) * 512);
            v16h b = pb[lane];
            c[j] = __builtin_amdgcn_wmma_f32_16x16x32_f16(false, a, false, b,
                                                          (short)0, c[j], false, false);
        }
    }
#pragma unroll
    for (int j = 0; j < 2; ++j) {
        int col = (wv * 2 + j) * 16 + nn;
        float bv = bias[col];
#pragma unroll
        for (int r = 0; r < 8; ++r) {
            int node = n0 + hw * 8 + r;
            size_t idx = (size_t)node * 128 + col;
            float v = lrelu(c[j][r] + bv) + x[idx];
            x[idx] = v;
            xh[idx] = (_Float16)v;
        }
    }
}

// ---------------- gate + per-graph softmax ----------------
__global__ void k_gate(const float* __restrict__ x, const float* __restrict__ gw,
                       const float* __restrict__ gb, float* __restrict__ gmax,
                       float* __restrict__ ge) {
    int n = blockIdx.x * 256 + threadIdx.x;
    const float4* xp = (const float4*)(x + (size_t)n * 128);
    const float4* wp = (const float4*)gw;
    float s = gb[0];
#pragma unroll
    for (int k = 0; k < 32; ++k) {
        float4 a = xp[k], b = wp[k];
        s += a.x*b.x + a.y*b.y + a.z*b.z + a.w*b.w;
    }
    ge[n] = s;
    atomicMaxF(&gmax[n >> 10], s);
}

__global__ void k_gate_exp(float* __restrict__ ge, const float* __restrict__ gmax,
                           float* __restrict__ gden) {
    int n = blockIdx.x * 256 + threadIdx.x;
    float e = __expf(ge[n] - gmax[n >> 10]);
    ge[n] = e;
    atomicAdd(&gden[n >> 10], e);
}

// ---------------- feat GEMM + attention pooling (WMMA) ----------------
__global__ void __launch_bounds__(128) k_featpool(
    const _Float16* __restrict__ xh, const _Float16* __restrict__ wfrag,
    const float* __restrict__ bias, const float* __restrict__ ge,
    const float* __restrict__ gden, float* __restrict__ xgacc) {
    __shared__ __align__(16) _Float16 lx[16 * 136];
    __shared__ float lw[16];
    int t = threadIdx.x;
    int n0 = blockIdx.x * 16;
    int g = n0 >> 10;
    if (t < 16) lw[t] = ge[n0 + t] / gden[g];
    {
        int row = t >> 3, seg = t & 7;
        async_copy32B(xh + (size_t)(n0 + row) * 128 + seg * 16,
                      &lx[row * 136 + seg * 16]);
    }
    async_wait0();
    __syncthreads();
    int lane = t & 31, wv = t >> 5, hw = lane >> 4, nn = lane & 15;
    v8f c[2] = {v8f{}, v8f{}};
#pragma unroll
    for (int kc = 0; kc < 4; ++kc) {
        v16h a = load_a_frag(&lx[nn * 136 + kc * 32 + hw * 8]);
#pragma unroll
        for (int j = 0; j < 2; ++j) {
            const v16h* pb = (const v16h*)(wfrag + (size_t)(kc * 8 + wv * 2 + j) * 512);
            v16h b = pb[lane];
            c[j] = __builtin_amdgcn_wmma_f32_16x16x32_f16(false, a, false, b,
                                                          (short)0, c[j], false, false);
        }
    }
#pragma unroll
    for (int j = 0; j < 2; ++j) {
        int col = (wv * 2 + j) * 16 + nn;
        float bv = bias[col];
        float part = 0.0f;
#pragma unroll
        for (int r = 0; r < 8; ++r) part += lw[hw * 8 + r] * lrelu(c[j][r] + bv);
        part += __shfl_xor(part, 16, 32);
        if (hw == 0) atomicAdd(&xgacc[(size_t)g * 128 + col], part);
    }
}

// ---------------- global transform (64x256x128, scalar) ----------------
__global__ void k_tr(const float* __restrict__ xgacc, const float* __restrict__ xgcur,
                     const float* __restrict__ W, const float* __restrict__ b,
                     float* __restrict__ xgnxt) {
    int g = blockIdx.x, c = threadIdx.x;
    float s = b[c];
    for (int k = 0; k < 128; ++k) s += xgacc[g * 128 + k] * W[(size_t)k * 128 + c];
    for (int k = 0; k < 128; ++k) s += xgcur[g * 128 + k] * W[(size_t)(128 + k) * 128 + c];
    xgnxt[g * 128 + c] = lrelu(s) + xgcur[g * 128 + c];
}

// ---------------- heads ----------------
__global__ void k_head_graph(const float* __restrict__ xg, const float* __restrict__ aW,
                             const float* __restrict__ ab, const float* __restrict__ vW,
                             const float* __restrict__ vb, float* __restrict__ out) {
    int g = threadIdx.x;
    float l[5];
#pragma unroll
    for (int j = 0; j < 5; ++j) l[j] = ab[j];
    float v = vb[0];
    for (int k = 0; k < 128; ++k) {
        float xv = xg[g * 128 + k];
        v += xv * vW[k];
#pragma unroll
        for (int j = 0; j < 5; ++j) l[j] += xv * aW[k * 5 + j];
    }
    float mx = l[0];
#pragma unroll
    for (int j = 1; j < 5; ++j) mx = fmaxf(mx, l[j]);
    float den = 0.0f, e[5];
#pragma unroll
    for (int j = 0; j < 5; ++j) { e[j] = __expf(l[j] - mx); den += e[j]; }
#pragma unroll
    for (int j = 0; j < 5; ++j) out[g * 5 + j] = e[j] / den;
    out[320 + 327680 + g] = v;
}

__global__ void k_node_init(float* __restrict__ nmax, float* __restrict__ nsum) {
    int t = threadIdx.x;
    nmax[t] = -__builtin_inff();
    nsum[t] = 0.0f;
}

__global__ void k_node_logits(const float* __restrict__ x, const float* __restrict__ W,
                              const float* __restrict__ b, float* __restrict__ nlog,
                              float* __restrict__ nmax) {
    int n = blockIdx.x * 128 + threadIdx.x;
    float a[5];
#pragma unroll
    for (int j = 0; j < 5; ++j) a[j] = b[j];
    const float* xr = x + (size_t)n * 128;
    for (int k = 0; k < 128; ++k) {
        float xv = xr[k];
        const float* w = W + k * 5;
#pragma unroll
        for (int j = 0; j < 5; ++j) a[j] += xv * w[j];
    }
    int g = n >> 10;
#pragma unroll
    for (int j = 0; j < 5; ++j) {
        nlog[(size_t)n * 5 + j] = a[j];
        atomicMaxF(&nmax[g * 5 + j], a[j]);
    }
}

__global__ void k_node_exp(float* __restrict__ nlog, const float* __restrict__ nmax,
                           float* __restrict__ nsum) {
    int id = blockIdx.x * 256 + threadIdx.x;     // < N*5
    int n = id / 5, j = id % 5, g = n >> 10;
    float e = __expf(nlog[id] - nmax[g * 5 + j]);
    nlog[id] = e;
    atomicAdd(&nsum[g * 5 + j], e);
}

__global__ void k_node_norm(const float* __restrict__ nlog, const float* __restrict__ nsum,
                            float* __restrict__ out) {
    int id = blockIdx.x * 256 + threadIdx.x;
    int n = id / 5, j = id % 5, g = n >> 10;
    out[id] = nlog[id] / nsum[g * 5 + j];
}

extern "C" void kernel_launch(void* const* d_in, const int* in_sizes, int n_in,
                              void* d_out, int out_size, void* d_ws, size_t ws_size,
                              hipStream_t stream) {
    (void)in_sizes; (void)n_in; (void)out_size; (void)ws_size;
    const float* node_feats = (const float*)d_in[0];
    const float* edge_attr  = (const float*)d_in[1];
    const int*   edge_index = (const int*)d_in[2];
    const float* embed_W = (const float*)d_in[4];
    const float* embed_b = (const float*)d_in[5];
    const float* mess_W  = (const float*)d_in[6];
    const float* mess_b  = (const float*)d_in[7];
    const float* agg_W   = (const float*)d_in[8];
    const float* agg_b   = (const float*)d_in[9];
    const float* gate_W  = (const float*)d_in[10];
    const float* gate_b  = (const float*)d_in[11];
    const float* feat_W  = (const float*)d_in[12];
    const float* feat_b  = (const float*)d_in[13];
    const float* tr_W    = (const float*)d_in[14];
    const float* tr_b    = (const float*)d_in[15];
    const float* node_W  = (const float*)d_in[16];
    const float* node_b  = (const float*)d_in[17];
    const float* act_W   = (const float*)d_in[18];
    const float* act_b   = (const float*)d_in[19];
    const float* val_W   = (const float*)d_in[20];
    const float* val_b   = (const float*)d_in[21];

    char* ws = (char*)d_ws;
    float*     x     = (float*)(ws + OFF_X);
    _Float16*  xh    = (_Float16*)(ws + OFF_XH);
    float*     agg   = (float*)(ws + OFF_AGG);
    float*     xgbuf = (float*)(ws + OFF_XG);
    _Float16*  xgh   = (_Float16*)(ws + OFF_XGH);
    float*     xgacc = (float*)(ws + OFF_XGACC);
    float*     gate  = (float*)(ws + OFF_GATE);
    float*     gmax  = (float*)(ws + OFF_GMAX);
    float*     gden  = (float*)(ws + OFF_GDEN);
    float*     nmax  = (float*)(ws + OFF_NMAX);
    float*     nsum  = (float*)(ws + OFF_NSUM);
    float*     nlog  = (float*)(ws + OFF_NLOG);
    _Float16*  frag  = (_Float16*)(ws + OFF_FRAG);
    float*     out   = (float*)d_out;

    const int* src = edge_index;
    const int* dst = edge_index + NE;

    k_prep<<<1280, 256, 0, stream>>>(mess_W, agg_W, feat_W, frag);
    k_zero_xg<<<64, 256, 0, stream>>>(xgbuf);
    k_embed<<<NN, 128, 0, stream>>>(node_feats, embed_W, embed_b, x, xh);

    for (int s = 0; s < NSTEPS; ++s) {
        int cur = s & 1, nxt = cur ^ 1;
        float* xg_cur = xgbuf + (size_t)cur * NG * DEMB;
        float* xg_nxt = xgbuf + (size_t)nxt * NG * DEMB;
        const _Float16* fr = frag + (size_t)s * FRAG_STEP_STRIDE;

        k_step_init<<<NN * DEMB / 256, 256, 0, stream>>>(agg, xgacc, xgh, xg_cur, gmax, gden);
        k_message<<<NE / 16, 128, 0, stream>>>(
            xh, edge_attr, src, dst, fr + FRAG_MESS_OFF,
            mess_W + (size_t)s * 132 * 128 + 128 * 128, mess_b + s * 128, agg);
        k_update<<<NN / 16, 128, 0, stream>>>(xh, xgh, agg, fr + FRAG_AGG_OFF,
                                              agg_b + s * 128, x);
        k_gate<<<NN / 256, 256, 0, stream>>>(x, gate_W + s * 128, gate_b + s, gmax, gate);
        k_gate_exp<<<NN / 256, 256, 0, stream>>>(gate, gmax, gden);
        k_featpool<<<NN / 16, 128, 0, stream>>>(xh, fr + FRAG_FEAT_OFF, feat_b + s * 128,
                                                gate, gden, xgacc);
        k_tr<<<NG, 128, 0, stream>>>(xgacc, xg_cur, tr_W + (size_t)s * 256 * 128,
                                     tr_b + s * 128, xg_nxt);
    }

    float* xg_fin = xgbuf;   // after 4 steps, result is in buffer 0
    k_head_graph<<<1, 64, 0, stream>>>(xg_fin, act_W, act_b, val_W, val_b, out);
    k_node_init<<<1, 320, 0, stream>>>(nmax, nsum);
    k_node_logits<<<NN / 128, 128, 0, stream>>>(x, node_W, node_b, nlog, nmax);
    k_node_exp<<<NN * 5 / 256, 256, 0, stream>>>(nlog, nmax, nsum);
    k_node_norm<<<NN * 5 / 256, 256, 0, stream>>>(nlog, nsum, out + 320);
}
